// EnhancedSNN_12154757448292
// MI455X (gfx1250) — matrix operand order repaired and checked by reading, not compiled, not run on hardware
//
#include <hip/hip_runtime.h>
#include <hip/hip_bf16.h>
#include <math.h>

typedef __attribute__((ext_vector_type(16))) _Float16 v16h;
typedef __attribute__((ext_vector_type(8)))  float    v8f;

#define MB     64     // rows per workgroup block
#define LDA    520    // half-element stride of big activation buffers (mult of 8)
#define NWAVE  8

// ---- LDS layout (bytes) ----
#define OFF_A    0         // 64*520*2 = 66560
#define OFF_B    66560     // 66560
#define OFF_RED  133120    // 64*8*4 = 2048
#define SMEM_BYTES 135168
// time-loop overlay (bufA/bufB are dead by then; x0 confined to bufB region)
#define OFF_M0   0
#define OFF_M1   16384
#define OFF_M2   32768
#define OFF_M3   40960
#define OFF_S0   49152     // 64x72 f16
#define OFF_X0   66560     // 64x64 f32 (inside bufB region, written in stage H)
#define OFF_S1   82944     // 64x72 f16
#define OFF_S2   92160     // 64x40 f16
#define OFF_S3   97280     // 64x40 f16
#define OFF_ACC  102400    // 64x8 f32

struct Params {
  const float *x, *g_in, *b_in, *b_vis, *g_vis, *bg_vis, *b_qkv_v, *b_o;
  const float *b1, *g1, *bg1, *b2, *g2, *bg2, *b3, *g3, *bg3;
  const float *bs0, *bs1, *bs2, *bs3, *b_out;
  const _Float16 *Wv, *Wqkv, *Wo, *W1, *W2, *W3, *Ws0, *Ws1, *Ws2, *Ws3, *Wout;
  float* out;
};

// ---------- weight prep: fp32 (K,N) row-major -> f16 (N,K) row-major ----------
__global__ void wtrans_kernel(const float* __restrict__ W, _Float16* __restrict__ out,
                              int K, int N, int ldw, int colOff) {
  int i = blockIdx.x * 256 + threadIdx.x;
  if (i >= K * N) return;
  int n = i / K, k = i - n * K;
  out[n * K + k] = (_Float16)W[(size_t)k * ldw + colOff + n];
}

// W_out (32,5) -> padded f16 (16,32): rows n>=5 zero
__global__ void woutpad_kernel(const float* __restrict__ W, _Float16* __restrict__ out) {
  int i = blockIdx.x * 256 + threadIdx.x;
  if (i >= 512) return;
  int n = i >> 5, k = i & 31;
  out[i] = (n < 5) ? (_Float16)W[k * 5 + n] : (_Float16)0.f;
}

// ---------- fragment loads ----------
__device__ __forceinline__ v16h ldfrag(const _Float16* p0, const _Float16* p1) {
  union { uint4 u[2]; v16h h; } t;
  t.u[0] = *(const uint4*)p0;
  t.u[1] = *(const uint4*)p1;
  return t.h;
}

// 16x16 tile, K-loop in steps of 32 with v_wmma_f32_16x16x32_f16
__device__ __forceinline__ v8f wmma_tile(const _Float16* aRow, const _Float16* bRow, int K) {
  v8f acc = {};
  for (int k0 = 0; k0 < K; k0 += 32) {
    v16h a = ldfrag(aRow + k0, aRow + k0 + 16);  // A: halves 0-7 at Kbase, 8-15 at Kbase+16
    v16h b = ldfrag(bRow + k0, bRow + k0 + 16);  // B: halves 0-7 at Kbase, 8-15 at Kbase+16
    acc = __builtin_amdgcn_wmma_f32_16x16x32_f16(false, a, false, b, (short)0, acc, false, false);
  }
  return acc;
}

// ---------- epilogue helper: bias (+optional lrelu), f16 column store ----------
__device__ __forceinline__ void epi_store_f16(v8f acc, float bv, _Float16* op,
                                              int mbase, int ldo, int lrelu) {
#pragma unroll
  for (int q = 0; q < 8; ++q) {
    float xv = acc[q] + bv;
    if (lrelu) xv = (xv >= 0.f) ? xv : 0.1f * xv;
    op[(mbase + q) * ldo] = (_Float16)xv;
  }
}

// ---------- 2x2 register-blocked GEMM (each wave owns a 32x32 output quad) ----
// A fragments reused across 2 N tiles, B fragments across 2 M tiles:
// 2 fragment loads per WMMA instead of 4.
__device__ void gemm_f16_2x2(const _Float16* A, int lda, const _Float16* Bt, int K, int N,
                             const float* bias, _Float16* out, int ldo, int lrelu,
                             int wave, int lane) {
  int nQ = N >> 5;               // quads along N
  int quads = (MB / 32) * nQ;    // MB=64 -> 2 row-pairs
  int rsel = lane >> 4, col = lane & 15;
  for (int t = wave; t < quads; t += NWAVE) {
    int mP = t / nQ, nP = t - mP * nQ;
    int m0 = mP * 32, n0 = nP * 32;
    const _Float16* aR0 = A + (m0 + col) * lda + rsel * 8;
    const _Float16* aR1 = aR0 + 16 * lda;
    const _Float16* bR0 = Bt + (n0 + col) * K + rsel * 8;
    const _Float16* bR1 = bR0 + 16 * K;
    v8f c00 = {}, c01 = {}, c10 = {}, c11 = {};
    for (int k0 = 0; k0 < K; k0 += 32) {
      v16h a0 = ldfrag(aR0 + k0, aR0 + k0 + 16);
      v16h a1 = ldfrag(aR1 + k0, aR1 + k0 + 16);
      v16h b0 = ldfrag(bR0 + k0, bR0 + k0 + 16);
      v16h b1 = ldfrag(bR1 + k0, bR1 + k0 + 16);
      c00 = __builtin_amdgcn_wmma_f32_16x16x32_f16(false, a0, false, b0, (short)0, c00, false, false);
      c01 = __builtin_amdgcn_wmma_f32_16x16x32_f16(false, a0, false, b1, (short)0, c01, false, false);
      c10 = __builtin_amdgcn_wmma_f32_16x16x32_f16(false, a1, false, b0, (short)0, c10, false, false);
      c11 = __builtin_amdgcn_wmma_f32_16x16x32_f16(false, a1, false, b1, (short)0, c11, false, false);
    }
    float bv0 = bias[n0 + col];
    float bv1 = bias[n0 + 16 + col];
    int mb0 = m0 + rsel * 8, mb1 = m0 + 16 + rsel * 8;
    epi_store_f16(c00, bv0, out + n0 + col,      mb0, ldo, lrelu);
    epi_store_f16(c01, bv1, out + n0 + 16 + col, mb0, ldo, lrelu);
    epi_store_f16(c10, bv0, out + n0 + col,      mb1, ldo, lrelu);
    epi_store_f16(c11, bv1, out + n0 + 16 + col, mb1, ldo, lrelu);
  }
}

// ---------- unblocked GEMM variants (small stages) ----------
__device__ void gemm_f16(const _Float16* A, int lda, const _Float16* Bt, int K, int N,
                         const float* bias, _Float16* out, int ldo, int lrelu,
                         int wave, int lane) {
  int nT = N >> 4;
  int tiles = (MB / 16) * nT;
  int rsel = lane >> 4, col = lane & 15;
  for (int t = wave; t < tiles; t += NWAVE) {
    int mT = t / nT, nTi = t - mT * nT;
    const _Float16* aRow = A + (mT * 16 + col) * lda + rsel * 8;
    const _Float16* bRow = Bt + (nTi * 16 + col) * K + rsel * 8;
    v8f acc = wmma_tile(aRow, bRow, K);
    epi_store_f16(acc, bias[nTi * 16 + col], out + nTi * 16 + col,
                  mT * 16 + rsel * 8, ldo, lrelu);
  }
}

__device__ void gemm_f32(const _Float16* A, int lda, const _Float16* Bt, int K, int N,
                         const float* bias, float* out, int ldo, int wave, int lane) {
  int nT = N >> 4;
  int tiles = (MB / 16) * nT;
  int rsel = lane >> 4, col = lane & 15;
  for (int t = wave; t < tiles; t += NWAVE) {
    int mT = t / nT, nTi = t - mT * nT;
    const _Float16* aRow = A + (mT * 16 + col) * lda + rsel * 8;
    const _Float16* bRow = Bt + (nTi * 16 + col) * K + rsel * 8;
    v8f acc = wmma_tile(aRow, bRow, K);
    float bv = bias[nTi * 16 + col];
    int mbase = mT * 16 + rsel * 8;
    float* op = out + nTi * 16 + col;
#pragma unroll
    for (int q = 0; q < 8; ++q) op[(mbase + q) * ldo] = acc[q] + bv;
  }
}

// GEMM with fused LIF epilogue: m state f32 (ld == N), spike f16 out
__device__ void gemm_lif(const _Float16* A, int lda, const _Float16* Bt, int K, int N,
                         const float* bias, float* m, _Float16* sOut, int ldso,
                         float beta, float thr, int wave, int lane) {
  int nT = N >> 4;
  int tiles = (MB / 16) * nT;
  int rsel = lane >> 4, col = lane & 15;
  for (int t = wave; t < tiles; t += NWAVE) {
    int mT = t / nT, nTi = t - mT * nT;
    const _Float16* aRow = A + (mT * 16 + col) * lda + rsel * 8;
    const _Float16* bRow = Bt + (nTi * 16 + col) * K + rsel * 8;
    v8f acc = wmma_tile(aRow, bRow, K);
    float bv = bias[nTi * 16 + col];
    int mbase = mT * 16 + rsel * 8;
    int nIdx = nTi * 16 + col;
#pragma unroll
    for (int q = 0; q < 8; ++q) {
      float u = acc[q] + bv;
      int idx = (mbase + q) * N + nIdx;
      float mm = m[idx];
      mm = (beta * mm + u) * ((mm > thr) ? 0.f : 1.f);
      m[idx] = mm;
      sOut[(mbase + q) * ldso + nIdx] = (_Float16)(((mm - thr) > 0.f) ? 1.f : 0.f);
    }
  }
}

// final GEMM (K=32, N padded to 16; only n<5 valid) with sigmoid accumulation
__device__ void gemm_sigacc(const _Float16* A, int lda, const _Float16* Bt,
                            const float* bias, float* accO, int wave, int lane) {
  int rsel = lane >> 4, col = lane & 15;
  for (int t = wave; t < 4; t += NWAVE) {
    const _Float16* aRow = A + (t * 16 + col) * lda + rsel * 8;
    const _Float16* bRow = Bt + col * 32 + rsel * 8;
    v8f acc = wmma_tile(aRow, bRow, 32);
    if (col < 5) {
      float bv = bias[col];
      int mbase = t * 16 + rsel * 8;
#pragma unroll
      for (int q = 0; q < 8; ++q) {
        float u = acc[q] + bv;
        accO[(mbase + q) * 8 + col] += 1.f / (1.f + expf(-u));
      }
    }
  }
}

// ---------- LayerNorm (in-place over f16 LDS rows) + optional residual -------
// vectorized: 8 halves (16B) per LDS access; per (=C/4) is always a multiple of 8
__device__ void ln_pass(_Float16* buf, int ld, int C, const float* g, const float* b,
                        const _Float16* res, int ldr, float* red, int tid) {
  int row = tid >> 2, part = tid & 3;
  int per = C >> 2;
  _Float16* p = buf + row * ld + part * per;
  float s = 0.f, ss = 0.f;
  for (int i = 0; i < per; i += 8) {
    union { uint4 u; _Float16 h[8]; } t;
    t.u = *(const uint4*)(p + i);
#pragma unroll
    for (int j = 0; j < 8; ++j) { float x = (float)t.h[j]; s += x; ss += x * x; }
  }
  red[row * 8 + part] = s;
  red[row * 8 + 4 + part] = ss;
  __syncthreads();
  float sum = 0.f, sq = 0.f;
#pragma unroll
  for (int j = 0; j < 4; ++j) { sum += red[row * 8 + j]; sq += red[row * 8 + 4 + j]; }
  float mean = sum / (float)C;
  float var = sq / (float)C - mean * mean;
  float r = rsqrtf(var + 1e-5f);
  const _Float16* rp = res ? (res + row * ldr + part * per) : (const _Float16*)0;
  for (int i = 0; i < per; i += 8) {
    union { uint4 u; _Float16 h[8]; } t, o, rr;
    t.u = *(const uint4*)(p + i);
    if (rp) rr.u = *(const uint4*)(rp + i);
#pragma unroll
    for (int j = 0; j < 8; ++j) {
      int n = part * per + i + j;
      float x = ((float)t.h[j] - mean) * r * g[n] + b[n];
      if (rp) x += (float)rr.h[j];
      o.h[j] = (_Float16)x;
    }
    *(uint4*)(p + i) = o.u;
  }
  __syncthreads();
}

// input LN: fp32 global (64x128) -> f16 LDS bufA
__device__ void input_ln(const float* xg, const float* g, const float* b,
                         _Float16* bufA, float* red, int tid) {
  int row = tid >> 2, part = tid & 3;
  const float* p = xg + row * 128 + part * 32;
  float vals[32];
  float s = 0.f, ss = 0.f;
#pragma unroll
  for (int i = 0; i < 32; ++i) { float x = p[i]; vals[i] = x; s += x; ss += x * x; }
  red[row * 8 + part] = s;
  red[row * 8 + 4 + part] = ss;
  __syncthreads();
  float sum = 0.f, sq = 0.f;
#pragma unroll
  for (int j = 0; j < 4; ++j) { sum += red[row * 8 + j]; sq += red[row * 8 + 4 + j]; }
  float mean = sum / 128.f;
  float var = sq / 128.f - mean * mean;
  float r = rsqrtf(var + 1e-5f);
  _Float16* q = bufA + row * LDA + part * 32;
  for (int i = 0; i < 32; i += 8) {
    union { uint4 u; _Float16 h[8]; } o;
#pragma unroll
    for (int j = 0; j < 8; ++j) {
      int n = part * 32 + i + j;
      o.h[j] = (_Float16)((vals[i + j] - mean) * r * g[n] + b[n]);
    }
    *(uint4*)(q + i) = o.u;
  }
  __syncthreads();
}

// ---------- fused SNN kernel: one workgroup = 64 rows ----------
__global__ __launch_bounds__(256) void snn_fused(Params P) {
  __shared__ __align__(16) char smem[SMEM_BYTES];
  int tid = threadIdx.x;
  int wave = tid >> 5, lane = tid & 31;
  _Float16* bufA = (_Float16*)(smem + OFF_A);
  _Float16* bufB = (_Float16*)(smem + OFF_B);
  float* red = (float*)(smem + OFF_RED);

  const float* xg = P.x + (size_t)blockIdx.x * MB * 128;
  input_ln(xg, P.g_in, P.b_in, bufA, red, tid);                       // xln -> bufA[:,0:128]

  gemm_f16_2x2(bufA, LDA, P.Wv, 128, 512, P.b_vis, bufB, LDA, 1, wave, lane);
  __syncthreads();
  ln_pass(bufB, LDA, 512, P.g_vis, P.bg_vis, (const _Float16*)0, 0, red, tid); // vis -> bufB

  gemm_f16_2x2(bufB, LDA, P.Wqkv, 512, 512, P.b_qkv_v, bufA, LDA, 0, wave, lane); // v -> bufA
  __syncthreads();
  gemm_f16_2x2(bufA, LDA, P.Wo, 512, 512, P.b_o, bufB, LDA, 0, wave, lane);       // attn -> bufB
  __syncthreads();

  gemm_f16_2x2(bufB, LDA, P.W1, 512, 256, P.b1, bufA, LDA, 1, wave, lane);
  __syncthreads();
  ln_pass(bufA, LDA, 256, P.g1, P.bg1, bufB, LDA, red, tid);          // res1 -> bufA (+attn[:,:256])

  gemm_f16_2x2(bufA, LDA, P.W2, 256, 128, P.b2, bufB, LDA, 1, wave, lane);
  __syncthreads();
  ln_pass(bufB, LDA, 128, P.g2, P.bg2, bufA, LDA, red, tid);          // res2 -> bufB (+res1[:,:128])

  gemm_f16(bufB, LDA, P.W3, 128, 64, P.b3, bufA, LDA, 1, wave, lane);
  __syncthreads();
  ln_pass(bufA, LDA, 64, P.g3, P.bg3, bufB, LDA, red, tid);           // res3 -> bufA (+res2[:,:64])

  float* x0 = (float*)(smem + OFF_X0);                                 // lives in dead bufB region
  gemm_f32(bufA, LDA, P.Ws0, 64, 64, P.bs0, x0, 64, wave, lane);
  __syncthreads();

  // ---- time loop: state overlays dead activation buffers ----
  float* m0 = (float*)(smem + OFF_M0);
  float* m1 = (float*)(smem + OFF_M1);
  float* m2 = (float*)(smem + OFF_M2);
  float* m3 = (float*)(smem + OFF_M3);
  _Float16* s0 = (_Float16*)(smem + OFF_S0);
  _Float16* s1 = (_Float16*)(smem + OFF_S1);
  _Float16* s2 = (_Float16*)(smem + OFF_S2);
  _Float16* s3 = (_Float16*)(smem + OFF_S3);
  float* accO = (float*)(smem + OFF_ACC);
  for (int i = tid; i < MB * 64; i += 256) { m0[i] = 0.f; m1[i] = 0.f; }
  for (int i = tid; i < MB * 32; i += 256) { m2[i] = 0.f; m3[i] = 0.f; }
  for (int i = tid; i < MB * 8;  i += 256) accO[i] = 0.f;
  __syncthreads();

  for (int t = 0; t < 10; ++t) {
    for (int i = tid; i < MB * 64; i += 256) {                         // LIF0 (elementwise)
      int r = i >> 6, c = i & 63;
      float mm = m0[i];
      mm = (0.95f * mm + x0[i]) * ((mm > 0.8f) ? 0.f : 1.f);
      m0[i] = mm;
      s0[r * 72 + c] = (_Float16)(((mm - 0.8f) > 0.f) ? 1.f : 0.f);
    }
    __syncthreads();
    gemm_lif(s0, 72, P.Ws1, 64, 64, P.bs1, m1, s1, 72, 0.92f, 0.85f, wave, lane);
    __syncthreads();
    gemm_lif(s1, 72, P.Ws2, 64, 32, P.bs2, m2, s2, 40, 0.90f, 0.90f, wave, lane);
    __syncthreads();
    gemm_lif(s2, 40, P.Ws3, 32, 32, P.bs3, m3, s3, 40, 0.85f, 0.95f, wave, lane);
    __syncthreads();
    gemm_sigacc(s3, 40, P.Wout, P.b_out, accO, wave, lane);
    __syncthreads();
  }

  float* outg = P.out + (size_t)blockIdx.x * MB * 5;
  for (int i = tid; i < MB * 5; i += 256) {
    int r = i / 5, c = i - r * 5;
    outg[i] = accO[r * 8 + c] * 0.1f;
  }
}

// ---------- launcher ----------
extern "C" void kernel_launch(void* const* d_in, const int* in_sizes, int n_in,
                              void* d_out, int out_size, void* d_ws, size_t ws_size,
                              hipStream_t stream) {
  (void)in_sizes; (void)n_in; (void)out_size; (void)ws_size;
  _Float16* ws = (_Float16*)d_ws;
  _Float16* Wv   = ws + 0;        // 512x128
  _Float16* Wqkv = ws + 65536;    // 512x512 (cols 1024..1535 of W_qkv)
  _Float16* Wo   = ws + 327680;   // 512x512
  _Float16* W1   = ws + 589824;   // 256x512
  _Float16* W2   = ws + 720896;   // 128x256
  _Float16* W3   = ws + 753664;   // 64x128
  _Float16* Ws0  = ws + 761856;   // 64x64
  _Float16* Ws1  = ws + 765952;   // 64x64
  _Float16* Ws2  = ws + 770048;   // 32x64
  _Float16* Ws3  = ws + 772096;   // 32x32
  _Float16* Wout = ws + 773120;   // 16x32 padded

  auto T = [&](const void* W, _Float16* o, int K, int N, int ldw, int off) {
    int total = K * N;
    hipLaunchKernelGGL(wtrans_kernel, dim3((total + 255) / 256), dim3(256), 0, stream,
                       (const float*)W, o, K, N, ldw, off);
  };
  T(d_in[3],  Wv,   128, 512, 512,  0);
  T(d_in[7],  Wqkv, 512, 512, 1536, 1024);
  T(d_in[9],  Wo,   512, 512, 512,  0);
  T(d_in[11], W1,   512, 256, 256,  0);
  T(d_in[15], W2,   256, 128, 128,  0);
  T(d_in[19], W3,   128, 64,  64,   0);
  T(d_in[23], Ws0,  64,  64,  64,   0);
  T(d_in[25], Ws1,  64,  64,  64,   0);
  T(d_in[27], Ws2,  64,  32,  32,   0);
  T(d_in[29], Ws3,  32,  32,  32,   0);
  hipLaunchKernelGGL(woutpad_kernel, dim3(2), dim3(256), 0, stream, (const float*)d_in[31], Wout);

  Params P;
  P.x      = (const float*)d_in[0];
  P.g_in   = (const float*)d_in[1];
  P.b_in   = (const float*)d_in[2];
  P.b_vis  = (const float*)d_in[4];
  P.g_vis  = (const float*)d_in[5];
  P.bg_vis = (const float*)d_in[6];
  P.b_qkv_v= (const float*)d_in[8] + 1024;
  P.b_o    = (const float*)d_in[10];
  P.b1     = (const float*)d_in[12];
  P.g1     = (const float*)d_in[13];
  P.bg1    = (const float*)d_in[14];
  P.b2     = (const float*)d_in[16];
  P.g2     = (const float*)d_in[17];
  P.bg2    = (const float*)d_in[18];
  P.b3     = (const float*)d_in[20];
  P.g3     = (const float*)d_in[21];
  P.bg3    = (const float*)d_in[22];
  P.bs0    = (const float*)d_in[24];
  P.bs1    = (const float*)d_in[26];
  P.bs2    = (const float*)d_in[28];
  P.bs3    = (const float*)d_in[30];
  P.b_out  = (const float*)d_in[32];
  P.Wv = Wv; P.Wqkv = Wqkv; P.Wo = Wo; P.W1 = W1; P.W2 = W2; P.W3 = W3;
  P.Ws0 = Ws0; P.Ws1 = Ws1; P.Ws2 = Ws2; P.Ws3 = Ws3; P.Wout = Wout;
  P.out = (float*)d_out;

  hipLaunchKernelGGL(snn_fused, dim3(32768 / MB), dim3(256), 0, stream, P);
}
